// CharDecoder_8761733283999
// MI455X (gfx1250) — matrix operand order repaired and checked by reading, not compiled, not run on hardware
//
#include <hip/hip_runtime.h>

// ---------------------------------------------------------------------------
// CharDecoder (3-layer GRU autoregressive decode) for MI455X / gfx1250.
// Math in bf16 WMMA (v_wmma_f32_16x16x32_bf16), f32 accumulation.
// w_hh gate tiles staged into LDS per block via the Tensor Data Mover
// (tensor_load_to_lds + s_wait_tensorcnt), shared by all 8 waves of a block.
//   n=8192, d_z=128, d_h=512, d_emb=128, V=64, L=3, max_len=80
// ---------------------------------------------------------------------------

typedef __attribute__((ext_vector_type(16))) __bf16 bf16x16;
typedef __attribute__((ext_vector_type(8)))  __bf16 bf16x8;
typedef __attribute__((ext_vector_type(8)))  float  f32x8;
typedef __attribute__((ext_vector_type(4)))  unsigned int u32x4;
typedef __attribute__((ext_vector_type(4)))  int    i32x4;
typedef __attribute__((ext_vector_type(8)))  int    i32x8;

#define D_H     512
#define D_EMB   128
#define VOC     64
#define MAXLEN  80
#define SOS_TOK 1
#define EOS_TOK 2
#define PAD_TOK 0

// therock-10.0 headers (amdgpu-toolchain) ship this header and use the
// 6-arg tensor_load_to_lds builtin; ROCm 7.2 uses the 5-arg form.
#if __has_include(<hip/amd_detail/amd_gfx1250_TDM.h>)
#define TDM_6ARG 1
#endif

// ---------------- Tensor Data Mover: 2D tile (16 rows x K elts, bf16) -------
// D# packing per CDNA5 ISA ch.8: group0 = count/lds_addr/global_addr/type,
// group1 = data_size, tensor dims, tile dims, dim0 stride. Groups 2/3 zero
// (2D tile). Issued by one wave; tracked with TENSORcnt.
__device__ __forceinline__ void tdm_load_tile_16xK(unsigned lds_byte_off,
                                                   const __bf16* gsrc,
                                                   unsigned Kelts) {
    unsigned long long ga = (unsigned long long)(uintptr_t)gsrc;
    u32x4 g0;
    g0.x = 1u;                                         // count=1, user mode
    g0.y = lds_byte_off;                               // D#.lds_addr (bytes)
    g0.z = (unsigned)(ga & 0xffffffffu);               // global_addr[31:0]
    g0.w = (unsigned)((ga >> 32) & 0x01ffffffu)        // global_addr[56:32]
         | (2u << 30);                                 // type = 2 ("image")
    i32x8 g1;
    const unsigned td0 = Kelts;   // tensor_dim0 (elements per row)
    const unsigned td1 = 16u;     // tensor_dim1 (rows)
    const unsigned tl0 = Kelts;   // tile_dim0  = whole row
    const unsigned tl1 = 16u;     // tile_dim1  = 16 rows
    const unsigned long long st0 = Kelts;  // tensor_dim0_stride
    g1[0] = (int)(1u << 16);                                   // mask=0, data_size=2B
    g1[1] = (int)((td0 & 0xffffu) << 16);                      // dim0[15:0]
    g1[2] = (int)((td0 >> 16) | ((td1 & 0xffffu) << 16));      // dim0[31:16]|dim1[15:0]
    g1[3] = (int)((td1 >> 16) | (tl0 << 16));                  // dim1[31:16]|tile0
    g1[4] = (int)(tl1);                                        // tile1 | tile2=0
    g1[5] = (int)(st0 & 0xffffffffu);                          // stride0[31:0]
    g1[6] = (int)(st0 >> 32);                                  // stride0[47:32]
    g1[7] = 0;
    i32x4 gz = {0, 0, 0, 0};
#if defined(TDM_6ARG)
    i32x8 gz8 = {0, 0, 0, 0, 0, 0, 0, 0};
    __builtin_amdgcn_tensor_load_to_lds(g0, g1, gz, gz, gz8, 0);
#else
    __builtin_amdgcn_tensor_load_to_lds(g0, g1, gz, gz, 0);
#endif
}

// ---------------- WMMA operand loaders (bf16, 16x16x32) ---------------------
// A (16x32, MxK): lane L holds row M=L&15; K 0..15 (lanes 0-15)
// or 16..31 (lanes 16-31); VGPR j packs K=2j (lo16), 2j+1 (hi16) which
// matches little-endian contiguous memory -> two 16B loads per lane.
__device__ __forceinline__ bf16x16 load_a_tile(const __bf16* rowptr, int kc, int lane) {
    const int koff = kc + ((lane & 16) ? 16 : 0);
    bf16x8 lo = *(const bf16x8*)(rowptr + koff);
    bf16x8 hi = *(const bf16x8*)(rowptr + koff + 8);
    bf16x16 a;
#pragma unroll
    for (int i = 0; i < 8; ++i) { a[i] = lo[i]; a[i + 8] = hi[i]; }
    return a;
}

// B (32x16, KxN) for out = x @ W^T: B[k][n] = W[n][k]. Lane L holds column
// N=L&15 with 16 consecutive K values (lanes 0-15: K=kc..kc+15, lanes 16-31:
// K=kc+16..kc+31) -> one contiguous 32B read from weight row (rowbase+N).
__device__ __forceinline__ bf16x16 load_b_tile(const __bf16* W, int rowbase, int K,
                                               int kc, int lane) {
    const int koff = kc + ((lane & 16) ? 16 : 0);
    return *(const bf16x16*)(W + (size_t)(rowbase + (lane & 15)) * K + koff);
}

// Same pattern but from the LDS-staged compact tile [gate][16 rows][D_H].
__device__ __forceinline__ bf16x16 load_b_lds(const __bf16* sm, int gate,
                                              int kc, int lane) {
    const int koff = kc + ((lane & 16) ? 16 : 0);
    return *(const bf16x16*)(sm + (size_t)(gate * 16 + (lane & 15)) * D_H + koff);
}

__device__ __forceinline__ f32x8 wmma_bf16(bf16x16 a, bf16x16 b, f32x8 c) {
    return __builtin_amdgcn_wmma_f32_16x16x32_bf16(
        /*neg_a=*/false, a, /*neg_b=*/false, b,
        /*c_mod=*/(short)0, c, /*reuse_a=*/false, /*reuse_b=*/false);
}

__device__ __forceinline__ float sigmoidf_dev(float x) {
    return 1.0f / (1.0f + __expf(-x));
}

// ---------------------- fp32 -> bf16 conversion -----------------------------
__global__ void cvt_f32_to_bf16(const float* __restrict__ src,
                                __bf16* __restrict__ dst, int count) {
    int i = blockIdx.x * 256 + threadIdx.x;
    if (i < count) dst[i] = (__bf16)src[i];
}

// ---------------------- decode state init -----------------------------------
__global__ void init_state(int* __restrict__ x_prev, int* __restrict__ eos,
                           int* __restrict__ xgen, int* __restrict__ seq_len, int n) {
    int i = blockIdx.x * 256 + threadIdx.x;
    if (i < n) {
        x_prev[i] = SOS_TOK;
        eos[i] = 0;
        seq_len[i] = MAXLEN;
        xgen[(size_t)i * MAXLEN] = SOS_TOK;
    }
}

// ---------------------- h0 = Z @ z2h_w^T + b ---------------------------------
__global__ __launch_bounds__(256) void h0_gemm(
    const __bf16* __restrict__ Zb, const __bf16* __restrict__ Wz,
    const float* __restrict__ bz,
    __bf16* __restrict__ H0, __bf16* __restrict__ H1, __bf16* __restrict__ H2) {
    const int wave = threadIdx.x >> 5;
    const int lane = threadIdx.x & 31;
    const int tile = blockIdx.x * 8 + wave;
    const int m0 = (tile >> 5) << 4;         // 512/16 = 32 N-tiles
    const int n0 = (tile & 31) << 4;

    f32x8 acc = {};
    const __bf16* aptr = Zb + (size_t)(m0 + (lane & 15)) * D_EMB;
#pragma unroll
    for (int kc = 0; kc < D_EMB; kc += 32) {
        bf16x16 a = load_a_tile(aptr, kc, lane);
        bf16x16 b = load_b_tile(Wz, n0, D_EMB, kc, lane);
        acc = wmma_bf16(a, b, acc);
    }

    const int ncol = n0 + (lane & 15);
    const int mbase = m0 + ((lane & 16) ? 8 : 0);
    const float bias = bz[ncol];
#pragma unroll
    for (int rr = 0; rr < 8; ++rr) {
        __bf16 h = (__bf16)(acc[rr] + bias);
        size_t idx = (size_t)(mbase + rr) * D_H + ncol;
        H0[idx] = h; H1[idx] = h; H2[idx] = h;
    }
}

// ---------------------- fused GRU layer step --------------------------------
// gates = x @ Wih^T + h @ Whh^T (+biases); r,z accumulate both GEMMs, i_n and
// h_n are kept separate (r scales h_n only). 8 waves/block all share the same
// n0 (16 hidden columns) and cover 128 consecutive rows, so the block's three
// 16xD_H w_hh gate tiles are staged once into LDS by the TDM and consumed by
// every wave; w_ih tiles are identical across waves -> WGP-cache hits.
__global__ __launch_bounds__(256) void gru_layer(
    const __bf16* __restrict__ Xsrc,       // [n, Kx] (layers 1..2) or unused
    const int*    __restrict__ tok,        // non-null for layer 0 (emb gather)
    const __bf16* __restrict__ embB,       // [V, D_EMB] bf16
    const int Kx,
    const __bf16* __restrict__ Wih,        // [3*D_H, Kx]
    const __bf16* __restrict__ Whh,        // [3*D_H, D_H]
    const __bf16* __restrict__ Hprev,      // [n, D_H]
    __bf16*       __restrict__ Hnext,      // [n, D_H]
    const float*  __restrict__ bih,        // [3*D_H]
    const float*  __restrict__ bhh) {      // [3*D_H]
    __shared__ __align__(16) __bf16 sWhh[3 * 16 * D_H];   // 48 KB

    const int wave = threadIdx.x >> 5;
    const int lane = threadIdx.x & 31;
    const int n0 = (blockIdx.x & 31) << 4;                 // shared by all waves
    const int m0 = (((blockIdx.x >> 5) << 3) + wave) << 4; // per-wave M tile

    // Stage the three w_hh gate tiles (rows n0 / n0+512 / n0+1024, 16xD_H)
    // into LDS with the Tensor Data Mover; one wave issues, all waves consume.
    if (wave == 0) {
        const unsigned base = (unsigned)(unsigned long long)(uintptr_t)&sWhh[0];
        tdm_load_tile_16xK(base + 0u * 16 * D_H * 2,
                           Whh + (size_t)(n0) * D_H, D_H);
        tdm_load_tile_16xK(base + 1u * 16 * D_H * 2,
                           Whh + (size_t)(n0 + D_H) * D_H, D_H);
        tdm_load_tile_16xK(base + 2u * 16 * D_H * 2,
                           Whh + (size_t)(n0 + 2 * D_H) * D_H, D_H);
        __builtin_amdgcn_s_wait_tensorcnt((short)0);
    }
    __syncthreads();

    f32x8 accR = {}, accZ = {}, accIN = {}, accHN = {};

    const int arow = m0 + (lane & 15);
    const __bf16* aptr = tok ? (embB + (size_t)tok[arow] * Kx)
                             : (Xsrc + (size_t)arow * Kx);

    // x contribution: r, z, i_n (w_ih from global; identical across waves)
    for (int kc = 0; kc < Kx; kc += 32) {
        bf16x16 a  = load_a_tile(aptr, kc, lane);
        bf16x16 br = load_b_tile(Wih, n0,            Kx, kc, lane);
        bf16x16 bz = load_b_tile(Wih, n0 + D_H,      Kx, kc, lane);
        bf16x16 bn = load_b_tile(Wih, n0 + 2 * D_H,  Kx, kc, lane);
        accR  = wmma_bf16(a, br, accR);
        accZ  = wmma_bf16(a, bz, accZ);
        accIN = wmma_bf16(a, bn, accIN);
    }
    // h contribution: r, z, h_n (w_hh from the LDS-staged tiles)
    const __bf16* hptr = Hprev + (size_t)arow * D_H;
    for (int kc = 0; kc < D_H; kc += 32) {
        bf16x16 a  = load_a_tile(hptr, kc, lane);
        bf16x16 br = load_b_lds(sWhh, 0, kc, lane);
        bf16x16 bz = load_b_lds(sWhh, 1, kc, lane);
        bf16x16 bn = load_b_lds(sWhh, 2, kc, lane);
        accR  = wmma_bf16(a, br, accR);
        accZ  = wmma_bf16(a, bz, accZ);
        accHN = wmma_bf16(a, bn, accHN);
    }

    // Elementwise GRU update on the D tile (VGPR rr -> M=rr(+8), lane -> N).
    const int ncol  = n0 + (lane & 15);
    const int mbase = m0 + ((lane & 16) ? 8 : 0);
    const float br_  = bih[ncol]           + bhh[ncol];
    const float bz_  = bih[D_H + ncol]     + bhh[D_H + ncol];
    const float bin_ = bih[2 * D_H + ncol];
    const float bhn_ = bhh[2 * D_H + ncol];
#pragma unroll
    for (int rr = 0; rr < 8; ++rr) {
        const int m = mbase + rr;
        float r = sigmoidf_dev(accR[rr] + br_);
        float z = sigmoidf_dev(accZ[rr] + bz_);
        float g = tanhf(accIN[rr] + bin_ + r * (accHN[rr] + bhn_));
        float hp = (float)Hprev[(size_t)m * D_H + ncol];
        Hnext[(size_t)m * D_H + ncol] = (__bf16)((1.0f - z) * g + z * hp);
    }
}

// ---------------------- classifier + argmax + EOS bookkeeping ----------------
// V=64, K=512: tiny GEMV per row; one block (64 threads) per sequence.
// argmax over logits == argmax over softmax; '>' keeps first max (jnp.argmax).
__global__ __launch_bounds__(64) void classifier_step(
    const __bf16* __restrict__ H2, const float* __restrict__ Wv,
    const float* __restrict__ bv,
    int* __restrict__ x_prev, int* __restrict__ eos,
    int* __restrict__ xgen, int* __restrict__ seq_len, const int t) {
    const int m = blockIdx.x;
    const int v = threadIdx.x;      // 0..63
    const __bf16* h = H2 + (size_t)m * D_H;
    const float*  w = Wv + (size_t)v * D_H;
    float acc = bv[v];
    for (int k = 0; k < D_H; k += 8) {
#pragma unroll
        for (int j = 0; j < 8; ++j) acc += (float)h[k + j] * w[k + j];
    }
    __shared__ float sval[VOC];
    sval[v] = acc;
    __syncthreads();
    if (v == 0) {
        int best = 0; float bestv = sval[0];
        for (int i = 1; i < VOC; ++i)
            if (sval[i] > bestv) { bestv = sval[i]; best = i; }
        const int e = eos[m];
        xgen[(size_t)m * MAXLEN + t] = e ? PAD_TOK : best;   // masked write
        if (!e && best == EOS_TOK) { eos[m] = 1; seq_len[m] = t + 1; }
        x_prev[m] = best;                                    // carry = raw argmax
    }
}

// ---------------------------------------------------------------------------
static inline size_t align256(size_t x) { return (x + 255) & ~(size_t)255; }

extern "C" void kernel_launch(void* const* d_in, const int* in_sizes, int n_in,
                              void* d_out, int out_size, void* d_ws, size_t ws_size,
                              hipStream_t stream) {
    (void)n_in; (void)out_size; (void)ws_size;
    const float* Z        = (const float*)d_in[0];
    const float* emb      = (const float*)d_in[1];
    const float* z2h_w    = (const float*)d_in[2];
    const float* z2h_b    = (const float*)d_in[3];
    const float* w_ih0    = (const float*)d_in[4];
    const float* w_ih_r   = (const float*)d_in[5];
    const float* w_hh     = (const float*)d_in[6];
    const float* b_ih     = (const float*)d_in[7];
    const float* b_hh     = (const float*)d_in[8];
    const float* h2v_w    = (const float*)d_in[9];
    const float* h2v_b    = (const float*)d_in[10];

    const int n = in_sizes[0] / 128;   // 8192

    // ---- workspace carve-up (bf16 weight copies + ping-pong H + state) ----
    char* w = (char*)d_ws; size_t off = 0;
    __bf16* embB  = (__bf16*)(w + off); off = align256(off + (size_t)VOC * D_EMB * 2);
    __bf16* Zb    = (__bf16*)(w + off); off = align256(off + (size_t)n * D_EMB * 2);
    __bf16* z2hB  = (__bf16*)(w + off); off = align256(off + (size_t)D_H * D_EMB * 2);
    __bf16* wih0B = (__bf16*)(w + off); off = align256(off + (size_t)3 * D_H * D_EMB * 2);
    __bf16* wihRB = (__bf16*)(w + off); off = align256(off + (size_t)2 * 3 * D_H * D_H * 2);
    __bf16* whhB  = (__bf16*)(w + off); off = align256(off + (size_t)3 * 3 * D_H * D_H * 2);
    __bf16* Hb    = (__bf16*)(w + off); off = align256(off + (size_t)6 * n * D_H * 2);
    int* x_prev   = (int*)(w + off);    off = align256(off + (size_t)n * 4);
    int* eos      = (int*)(w + off);    off = align256(off + (size_t)n * 4);

    int* xgen    = (int*)d_out;                 // [n, MAXLEN] int32
    int* seq_len = (int*)d_out + (size_t)n * MAXLEN;

    auto Hbuf = [&](int layer, int par) -> __bf16* {
        return Hb + ((size_t)(layer * 2 + par)) * n * D_H;
    };

    // ---- weight / input conversions to bf16 (deterministic, every call) ----
    auto cvt = [&](const float* s, __bf16* d, int count) {
        cvt_f32_to_bf16<<<(count + 255) / 256, 256, 0, stream>>>(s, d, count);
    };
    cvt(emb,    embB,  VOC * D_EMB);
    cvt(Z,      Zb,    n * D_EMB);
    cvt(z2h_w,  z2hB,  D_H * D_EMB);
    cvt(w_ih0,  wih0B, 3 * D_H * D_EMB);
    cvt(w_ih_r, wihRB, 2 * 3 * D_H * D_H);
    cvt(w_hh,   whhB,  3 * 3 * D_H * D_H);

    const int tiles  = (n / 16) * (D_H / 16);   // 16384
    const int blocks = tiles / 8;               // 8 waves / block

    // h0 into parity-0 buffers of all 3 layers
    h0_gemm<<<blocks, 256, 0, stream>>>(Zb, z2hB, z2h_b,
                                        Hbuf(0, 0), Hbuf(1, 0), Hbuf(2, 0));
    init_state<<<(n + 255) / 256, 256, 0, stream>>>(x_prev, eos, xgen, seq_len, n);

    const size_t wg = (size_t)3 * D_H * D_H;    // per-layer w_hh/w_ih stride
    for (int t = 1; t < MAXLEN; ++t) {
        const int rb = (t - 1) & 1, wb = t & 1;
        // layer 0: x = emb[x_prev], Kx = 128
        gru_layer<<<blocks, 256, 0, stream>>>(
            nullptr, x_prev, embB, D_EMB,
            wih0B, whhB, Hbuf(0, rb), Hbuf(0, wb), b_ih, b_hh);
        // layer 1: x = h_new(layer0), Kx = 512
        gru_layer<<<blocks, 256, 0, stream>>>(
            Hbuf(0, wb), nullptr, nullptr, D_H,
            wihRB, whhB + wg, Hbuf(1, rb), Hbuf(1, wb),
            b_ih + 3 * D_H, b_hh + 3 * D_H);
        // layer 2
        gru_layer<<<blocks, 256, 0, stream>>>(
            Hbuf(1, wb), nullptr, nullptr, D_H,
            wihRB + wg, whhB + 2 * wg, Hbuf(2, rb), Hbuf(2, wb),
            b_ih + 6 * D_H, b_hh + 6 * D_H);
        // vocab projection + argmax + EOS/seq_len update
        classifier_step<<<n, VOC, 0, stream>>>(
            Hbuf(2, wb), h2v_w, h2v_b, x_prev, eos, xgen, seq_len, t);
    }
}
